// PointsToImage_89438398972576
// MI455X (gfx1250) — matrix operand order repaired and checked by reading, not compiled, not run on hardware
//
#include <hip/hip_runtime.h>
#include <hip/hip_bf16.h>

// ---------------------------------------------------------------------------
// Problem constants (match reference)
// ---------------------------------------------------------------------------
#define BB   32          // batch
#define NN   32          // image dim
#define PP   1024        // points per sample (N*N)
#define NP   32768       // B*P
#define FF   64          // feature dim
#define KK   32          // max neighbors
#define R2   0.01f       // radius^2
#define EPSV 1e-5f
#define H1   128         // hidden dim
#define KPAD 96          // 66 padded to 3*32 for bf16 WMMA k-steps
#define NB   (NP * KK)   // neighbor-edge rows = 1,048,576
#define MT   (NB + NP)   // total message rows = 1,081,344 (divisible by 64)
#define TM   64          // message rows per block tile

// LDS strides padded for conflict-free wave access on 64x4B banks:
//  - 104 ushort = 52 dwords: n*52 mod 64 distinct for n=0..15
//  - 136 ushort = 68 dwords: n*68 mod 64 = 4n distinct for n=0..15
#define MS   104         // msg row stride (>= KPAD)
#define W1S  104         // w1t row stride (>= KPAD)
#define H2S  136         // H row / w2t row stride (>= H1)

typedef __bf16 v16bf __attribute__((ext_vector_type(16)));
typedef float  v8f   __attribute__((ext_vector_type(8)));

union BF16x16 { v16bf v; unsigned short u[16]; };

#define DEV __device__ __forceinline__

DEV unsigned short f2bf(float f) {           // round-to-nearest-even f32 -> bf16
    unsigned u = __float_as_uint(f);
    u += 0x7FFFu + ((u >> 16) & 1u);
    return (unsigned short)(u >> 16);
}
DEV unsigned encf(float f) {                 // monotone float -> uint for atomicMax
    unsigned u = __float_as_uint(f);
    return (u & 0x80000000u) ? ~u : (u | 0x80000000u);
}
DEV float decf(unsigned e) {
    return (e & 0x80000000u) ? __uint_as_float(e ^ 0x80000000u) : __uint_as_float(~e);
}
#define ENC_NEG_INF 0x007FFFFFu              // encf(-inf)

// ---- WMMA fragment loaders (contiguous k order k = kbase + 16*half + e, same in A and B) ----
// A (16x32, MxK) from row-major [row][k]:
//   lane<16: row=lane, K=kbase..+15; lane>=16: K=kbase+16..+31
//   -> one contiguous 32B run per lane (matches load_B_fragT k order)
DEV v16bf load_A_frag(const unsigned short* lds, int ld, int rowbase, int kbase, int lane) {
    int half = (lane >> 4) & 1, r = rowbase + (lane & 15);
    BF16x16 f;
#pragma unroll
    for (int e = 0; e < 16; ++e) {
        int k = kbase + half * 16 + e;
        f.u[e] = lds[r * ld + k];
    }
    return f.v;
}
// B (32x16, KxN) from TRANSPOSED LDS [n][k]:
//   lane<16: col n=lane, K=kbase..+15; lane>=16: col n=lane-16, K=kbase+16..+31
//   -> one contiguous 32B run per lane (vectorizes to wide ds loads)
DEV v16bf load_B_fragT(const unsigned short* ldsT, int ldT, int kbase, int nbase, int lane) {
    int half = (lane >> 4) & 1, n = nbase + (lane & 15);
    const unsigned short* p = ldsT + n * ldT + kbase + half * 16;
    BF16x16 f;
#pragma unroll
    for (int e = 0; e < 16; ++e) f.u[e] = p[e];
    return f.v;
}

// ---------------------------------------------------------------------------
// Kernel 1: pixel index of every point
// ---------------------------------------------------------------------------
__global__ void pix_kernel(const float* __restrict__ pos, const int* __restrict__ batch,
                           int* __restrict__ pix) {
    int i = blockIdx.x * blockDim.x + threadIdx.x;
    if (i >= NP) return;
    int col = (int)(pos[i * 2 + 0] * NN); col = min(max(col, 0), NN - 1);
    int row = (int)(pos[i * 2 + 1] * NN); row = min(max(row, 0), NN - 1);
    pix[i] = batch[i] * (NN * NN) + row * NN + col;
}

// ---------------------------------------------------------------------------
// Kernel 2: prep — W1 -> bf16 TRANSPOSED [128][104] (k<66 valid, zero-pad),
//           W2 -> bf16 TRANSPOSED [64][136]; zero batchnorm accumulators
// ---------------------------------------------------------------------------
__global__ void prep_kernel(const float* __restrict__ W1, const float* __restrict__ W2,
                            unsigned short* __restrict__ w1t, unsigned short* __restrict__ w2t,
                            float* __restrict__ gsum, float* __restrict__ gsumsq,
                            float* __restrict__ gcnt) {
    int i = blockIdx.x * blockDim.x + threadIdx.x;
    if (i < H1 * W1S) {
        int n = i / W1S, k = i % W1S;
        w1t[i] = f2bf(k < 66 ? W1[k * H1 + n] : 0.0f);
    }
    if (i < FF * H2S) {
        int n = i / H2S, k = i % H2S;
        w2t[i] = f2bf(k < H1 ? W2[k * FF + n] : 0.0f);
    }
    if (i < H1) { gsum[i] = 0.0f; gsumsq[i] = 0.0f; }
    if (i == 0) gcnt[0] = 0.0f;
}

// ---------------------------------------------------------------------------
// Kernel 3: init scatter-max accumulator to enc(-inf)
// ---------------------------------------------------------------------------
__global__ void init_enc_kernel(unsigned int* __restrict__ enc) {
    int i = blockIdx.x * blockDim.x + threadIdx.x;
    if (i < NP * FF) enc[i] = ENC_NEG_INF;
}

// ---------------------------------------------------------------------------
// Kernel 4: ball query. One wave32 per query point; sample positions in LDS.
// 32 lexicographic-min selection rounds reproduce top_k(-d2) tie-breaking.
// ---------------------------------------------------------------------------
__global__ void ball_kernel(const float* __restrict__ pos, int* __restrict__ nbr) {
    __shared__ float sx[PP], sy[PP];
    const int b    = blockIdx.x / (PP / 8);
    const int qblk = blockIdx.x % (PP / 8);
    const int tid  = threadIdx.x;
    for (int i = tid; i < PP; i += 256) {
        sx[i] = pos[(b * PP + i) * 2 + 0];
        sy[i] = pos[(b * PP + i) * 2 + 1];
    }
    __syncthreads();

    const int wv = tid >> 5, lane = tid & 31;
    const int qi = qblk * 8 + wv;
    const float qx = sx[qi], qy = sy[qi];

    float d2reg[32];                       // candidate j = t*32 + lane (static-indexed only)
#pragma unroll
    for (int t = 0; t < 32; ++t) {
        int j = t * 32 + lane;
        float dx = sx[j] - qx, dy = sy[j] - qy;
        d2reg[t] = dx * dx + dy * dy;
    }

    float lastD = -1.0f; int lastJ = -1;
    const long long gq = (long long)b * PP + qi;
    for (int k = 0; k < KK; ++k) {
        float bd = 3.0e38f; int bj = 0x7FFFFFFF;
#pragma unroll
        for (int t = 0; t < 32; ++t) {
            int j = t * 32 + lane;
            float d = d2reg[t];
            bool ok = (d <= R2) && ((d > lastD) || (d == lastD && j > lastJ));
            if (ok && ((d < bd) || (d == bd && j < bj))) { bd = d; bj = j; }
        }
#pragma unroll
        for (int off = 16; off >= 1; off >>= 1) {
            float od = __shfl_xor(bd, off, 32);
            int   oj = __shfl_xor(bj, off, 32);
            if ((od < bd) || (od == bd && oj < bj)) { bd = od; bj = oj; }
        }
        if (lane == 0)
            nbr[gq * KK + k] = (bd <= R2) ? (b * PP + bj) : -1;
        lastD = bd; lastJ = bj;
    }
}

// ---------------------------------------------------------------------------
// Message-tile builder (shared by both MLP passes): 64 rows x 96 bf16 cols
// (stride MS), gathers x[src] + relpos (pos[src]-pos[pix]), self rows appended.
// ---------------------------------------------------------------------------
DEV void build_msg_tile(unsigned short* __restrict__ msg, int* __restrict__ tgt,
                        int* __restrict__ msk,
                        const float* __restrict__ x, const float* __restrict__ pos,
                        const int* __restrict__ pix, const int* __restrict__ nbr,
                        long long m0, int tid) {
    int r = tid >> 2, g = tid & 3;         // 4 threads per row, 24 cols each
    long long gm = m0 + r;
    int src, tg; bool valid; bool self = (gm >= (long long)NB);
    if (!self) {
        int q  = (int)(gm / KK);
        int nb = nbr[gm];                  // global node index or -1
        int pq = pix[q];
        src = nb; tg = pq;
        valid = (nb >= 0) && (nb != pq);   // remove_self_loops on (src, pix)
    } else {
        int i = (int)(gm - NB);
        src = i; tg = i; valid = true;
    }
    if (g == 0) { tgt[r] = tg; msk[r] = valid ? 1 : 0; }
    if (valid) __builtin_prefetch(&x[src * FF + g * 24], 0, 3);   // global_prefetch_b8
    for (int c = g * 24; c < g * 24 + 24; ++c) {
        float v = 0.0f;
        if (valid) {
            if (c < FF)        v = x[src * FF + c];
            else if (c == 64)  v = self ? 0.0f : (pos[src * 2 + 0] - pos[tg * 2 + 0]);
            else if (c == 65)  v = self ? 0.0f : (pos[src * 2 + 1] - pos[tg * 2 + 1]);
        }
        msg[r * MS + c] = f2bf(v);
    }
}

DEV v8f gemm1_tile(const unsigned short* __restrict__ msg, const unsigned short* __restrict__ w1t,
                   int rt, int ct, int lane) {
    v8f acc = {};
#pragma unroll
    for (int ks = 0; ks < 3; ++ks) {       // K = 96 = 3 x 32
        v16bf a = load_A_frag(msg, MS, rt * 16, ks * 32, lane);
        v16bf b = load_B_fragT(w1t, W1S, ks * 32, ct * 16, lane);
        acc = __builtin_amdgcn_wmma_f32_16x16x32_bf16(false, a, false, b, (short)0, acc,
                                                      false, false);
    }
    return acc;
}

// ---------------------------------------------------------------------------
// Kernel 5 (pass 1): GEMM1 tiles -> per-column sum / sumsq / count for BN.
// z never hits HBM (553 MB avoided; recompute is cheap against WMMA peak).
// ---------------------------------------------------------------------------
__global__ void stats_kernel(const float* __restrict__ x, const float* __restrict__ pos,
                             const int* __restrict__ pix, const int* __restrict__ nbr,
                             const float* __restrict__ b1,
                             const unsigned short* __restrict__ w1g,
                             float* __restrict__ gsum, float* __restrict__ gsumsq,
                             float* __restrict__ gcnt) {
    extern __shared__ char smem[];
    unsigned short* msg = (unsigned short*)smem;            // 64 x MS
    unsigned short* w1t = msg + TM * MS;                    // 128 x W1S (transposed)
    float*          Z   = (float*)(w1t + H1 * W1S);         // 64 x 128
    int*            tgt = (int*)(Z + TM * H1);
    int*            msk = tgt + TM;

    const int tid = threadIdx.x;
    for (int i = tid; i < H1 * W1S; i += 256) w1t[i] = w1g[i];
    build_msg_tile(msg, tgt, msk, x, pos, pix, nbr, (long long)blockIdx.x * TM, tid);
    __syncthreads();

    const int wv = tid >> 5, lane = tid & 31, half = (lane >> 4) & 1;
#pragma unroll
    for (int t = 0; t < 4; ++t) {          // 32 output tiles (4x8) over 8 waves
        int id = wv * 4 + t, rt = id >> 3, ct = id & 7;
        v8f acc = gemm1_tile(msg, w1t, rt, ct, lane);
        int n = ct * 16 + (lane & 15);
        float bb = b1[n];
#pragma unroll
        for (int r = 0; r < 8; ++r)
            Z[(rt * 16 + half * 8 + r) * H1 + n] = acc[r] + bb;
    }
    __syncthreads();

    if (tid < H1) {
        float s = 0.0f, s2 = 0.0f;
        for (int r = 0; r < TM; ++r)
            if (msk[r]) { float z = Z[r * H1 + tid]; s += z; s2 += z * z; }
        atomicAdd(&gsum[tid], s);
        atomicAdd(&gsumsq[tid], s2);
    }
    if (tid == 0) {
        int c = 0;
        for (int r = 0; r < TM; ++r) c += msk[r];
        atomicAdd(gcnt, (float)c);
    }
}

// ---------------------------------------------------------------------------
// Kernel 6: fold BN stats into affine (h = z*a + c)
// ---------------------------------------------------------------------------
__global__ void finalize_stats_kernel(const float* __restrict__ gsum,
                                      const float* __restrict__ gsumsq,
                                      const float* __restrict__ gcnt,
                                      const float* __restrict__ gamma1,
                                      const float* __restrict__ beta1,
                                      float* __restrict__ anorm, float* __restrict__ cnorm) {
    int c = threadIdx.x;
    if (c >= H1) return;
    float cnt  = gcnt[0];
    float mean = gsum[c] / cnt;
    float var  = gsumsq[c] / cnt - mean * mean;
    float a    = gamma1[c] * rsqrtf(var + EPSV);
    anorm[c] = a;
    cnorm[c] = beta1[c] - mean * a;
}

// ---------------------------------------------------------------------------
// Kernel 7 (pass 2): recompute GEMM1 -> BN+ReLU -> GEMM2 -> scatter-max,
// fully fused in LDS; WMMA for both GEMMs; atomics stay in L2.
// ---------------------------------------------------------------------------
__global__ void mlp_kernel(const float* __restrict__ x, const float* __restrict__ pos,
                           const int* __restrict__ pix, const int* __restrict__ nbr,
                           const float* __restrict__ b1, const float* __restrict__ b2,
                           const unsigned short* __restrict__ w1g,
                           const unsigned short* __restrict__ w2g,
                           const float* __restrict__ anorm, const float* __restrict__ cnorm,
                           unsigned int* __restrict__ enc) {
    extern __shared__ char smem[];
    unsigned short* msg = (unsigned short*)smem;            // 64 x MS
    unsigned short* w1t = msg + TM * MS;                    // 128 x W1S (transposed)
    unsigned short* w2t = w1t + H1 * W1S;                   // 64 x H2S  (transposed)
    unsigned short* H   = w2t + FF * H2S;                   // 64 x H2S  (bf16, padded)
    float*          O   = (float*)(H + TM * H2S);           // 64 x 64
    int*            tgt = (int*)(O + TM * FF);
    int*            msk = tgt + TM;

    const int tid = threadIdx.x;
    for (int i = tid; i < H1 * W1S; i += 256) w1t[i] = w1g[i];
    for (int i = tid; i < FF * H2S; i += 256) w2t[i] = w2g[i];
    build_msg_tile(msg, tgt, msk, x, pos, pix, nbr, (long long)blockIdx.x * TM, tid);
    __syncthreads();

    const int wv = tid >> 5, lane = tid & 31, half = (lane >> 4) & 1;

    // GEMM1 + BN + ReLU -> H (bf16)
#pragma unroll
    for (int t = 0; t < 4; ++t) {
        int id = wv * 4 + t, rt = id >> 3, ct = id & 7;
        v8f acc = gemm1_tile(msg, w1t, rt, ct, lane);
        int n = ct * 16 + (lane & 15);
        float bb = b1[n], aa = anorm[n], cc = cnorm[n];
#pragma unroll
        for (int r = 0; r < 8; ++r) {
            float h = fmaxf(0.0f, (acc[r] + bb) * aa + cc);
            H[(rt * 16 + half * 8 + r) * H2S + n] = f2bf(h);
        }
    }
    __syncthreads();

    // GEMM2: H[64,128] @ W2[128,64] -> O
#pragma unroll
    for (int t = 0; t < 2; ++t) {          // 16 tiles (4x4) over 8 waves
        int id = wv * 2 + t, rt = id >> 2, ct = id & 3;
        v8f acc = {};
#pragma unroll
        for (int ks = 0; ks < 4; ++ks) {   // K = 128 = 4 x 32
            v16bf a = load_A_frag(H, H2S, rt * 16, ks * 32, lane);
            v16bf b = load_B_fragT(w2t, H2S, ks * 32, ct * 16, lane);
            acc = __builtin_amdgcn_wmma_f32_16x16x32_bf16(false, a, false, b, (short)0, acc,
                                                          false, false);
        }
        int n = ct * 16 + (lane & 15);
        float bb = b2[n];
#pragma unroll
        for (int r = 0; r < 8; ++r)
            O[(rt * 16 + half * 8 + r) * FF + n] = acc[r] + bb;
    }
    __syncthreads();

    // scatter-max (monotone-uint encoded; stays in L2)
    for (int e = tid; e < TM * FF; e += 256) {
        int r = e >> 6, c = e & 63;
        if (msk[r])
            atomicMax(&enc[(long long)tgt[r] * FF + c], encf(O[e]));
    }
}

// ---------------------------------------------------------------------------
// Kernel 8: decode + transpose to [B, F, N, N]
// ---------------------------------------------------------------------------
__global__ void decode_kernel(const unsigned int* __restrict__ enc, float* __restrict__ out) {
    int i = blockIdx.x * blockDim.x + threadIdx.x;   // i = s*64 + f
    if (i >= NP * FF) return;
    int s = i >> 6, f = i & 63;
    int b = s >> 10, rc = s & 1023;
    out[((b * FF + f) << 10) + rc] = decf(enc[i]);
}

// ---------------------------------------------------------------------------
// Host launcher
// ---------------------------------------------------------------------------
static inline size_t alignUp(size_t v, size_t a) { return (v + a - 1) / a * a; }

extern "C" void kernel_launch(void* const* d_in, const int* in_sizes, int n_in,
                              void* d_out, int out_size, void* d_ws, size_t ws_size,
                              hipStream_t stream) {
    const float* x     = (const float*)d_in[0];
    const float* pos   = (const float*)d_in[1];
    const int*   batch = (const int*)d_in[2];
    const float* W1    = (const float*)d_in[3];
    const float* b1    = (const float*)d_in[4];
    const float* g1    = (const float*)d_in[5];
    const float* be1   = (const float*)d_in[6];
    const float* W2    = (const float*)d_in[7];
    const float* b2    = (const float*)d_in[8];
    float* out = (float*)d_out;

    // workspace carve-out
    char* w = (char*)d_ws;
    size_t off = 0;
    int* pix            = (int*)(w + off);            off = alignUp(off + NP * 4, 256);
    int* nbr            = (int*)(w + off);            off = alignUp(off + (size_t)NB * 4, 256);
    unsigned short* w1t = (unsigned short*)(w + off); off = alignUp(off + H1 * W1S * 2, 256);
    unsigned short* w2t = (unsigned short*)(w + off); off = alignUp(off + FF * H2S * 2, 256);
    float* gsum         = (float*)(w + off);          off = alignUp(off + H1 * 4, 256);
    float* gsumsq       = (float*)(w + off);          off = alignUp(off + H1 * 4, 256);
    float* gcnt         = (float*)(w + off);          off = alignUp(off + 4, 256);
    float* anorm        = (float*)(w + off);          off = alignUp(off + H1 * 4, 256);
    float* cnorm        = (float*)(w + off);          off = alignUp(off + H1 * 4, 256);
    unsigned int* enc   = (unsigned int*)(w + off);   off = alignUp(off + (size_t)NP * FF * 4, 256);

    prep_kernel<<<(H1 * W1S + 255) / 256, 256, 0, stream>>>(W1, W2, w1t, w2t,
                                                            gsum, gsumsq, gcnt);
    pix_kernel<<<(NP + 255) / 256, 256, 0, stream>>>(pos, batch, pix);
    init_enc_kernel<<<(NP * FF + 255) / 256, 256, 0, stream>>>(enc);
    ball_kernel<<<BB * (PP / 8), 256, 0, stream>>>(pos, nbr);

    const size_t lds1 = TM * MS * 2 + H1 * W1S * 2 + TM * H1 * 4 + TM * 8;      // ~73 KB
    stats_kernel<<<MT / TM, 256, lds1, stream>>>(x, pos, pix, nbr, b1, w1t,
                                                 gsum, gsumsq, gcnt);

    finalize_stats_kernel<<<1, H1, 0, stream>>>(gsum, gsumsq, gcnt, g1, be1, anorm, cnorm);

    const size_t lds2 = TM * MS * 2 + H1 * W1S * 2 + FF * H2S * 2
                      + TM * H2S * 2 + TM * FF * 4 + TM * 8;                     // ~92 KB
    mlp_kernel<<<MT / TM, 256, lds2, stream>>>(x, pos, pix, nbr, b1, b2, w1t, w2t,
                                               anorm, cnorm, enc);

    decode_kernel<<<(NP * FF + 255) / 256, 256, 0, stream>>>(enc, out);
}
